// GCN_2903397893032
// MI455X (gfx1250) — matrix operand order repaired and checked by reading, not compiled, not run on hardware
//
#include <hip/hip_runtime.h>

typedef __attribute__((ext_vector_type(16))) _Float16 v16h;
typedef __attribute__((ext_vector_type(8)))  _Float16 v8h;
typedef __attribute__((ext_vector_type(4)))  _Float16 h4;
typedef __attribute__((ext_vector_type(8)))  float    v8f;
typedef __attribute__((ext_vector_type(4)))  float    f4v;

#define NN 8192
#define FF 128
#define UU 128

#define CAT16(lo, hi) __builtin_shufflevector(lo, hi, 0,1,2,3,4,5,6,7,8,9,10,11,12,13,14,15)

// ---------------------------------------------------------------------------
// Kernel 1: dinv[i] = rsqrt(1 + sum_j adj[i][j])   (streams adj once, NT hint)
// ---------------------------------------------------------------------------
__global__ __launch_bounds__(256) void gcn_deg_kernel(
    const float* __restrict__ adj, float* __restrict__ dinv)
{
  __shared__ float red[256];
  const size_t row = blockIdx.x;
  const f4v* r4 = (const f4v*)(adj + row * (size_t)NN);
  float s = 0.f;
#pragma unroll 4
  for (int j = threadIdx.x; j < NN / 4; j += 256) {
    f4v v = __builtin_nontemporal_load(&r4[j]);
    s += v.x + v.y + v.z + v.w;
  }
  red[threadIdx.x] = s;
  __syncthreads();
  for (int off = 128; off > 0; off >>= 1) {
    if (threadIdx.x < off) red[threadIdx.x] += red[threadIdx.x + off];
    __syncthreads();
  }
  if (threadIdx.x == 0) dinv[row] = rsqrtf(red[0] + 1.0f);
}

// ---------------------------------------------------------------------------
// Kernel 2: xs_t[c][i] = f16(x[i][c] * dinv[i])  -- tiled transpose via LDS
// 128 blocks, each handles 64 rows of x.
// ---------------------------------------------------------------------------
__global__ __launch_bounds__(256) void gcn_prep_x_kernel(
    const float* __restrict__ x, const float* __restrict__ dinv,
    _Float16* __restrict__ xs_t)
{
  __shared__ _Float16 tile[128][72];        // [c][r], padded
  const int i0 = blockIdx.x * 64;
  // coalesced read + scale + scatter into LDS
  for (int idx = threadIdx.x; idx < (64 * FF) / 4; idx += 256) {
    int r  = idx >> 5;              // 0..63
    int c4 = (idx & 31) << 2;       // 0,4,...,124
    f4v v = *(const f4v*)(x + (size_t)(i0 + r) * FF + c4);
    float dv = dinv[i0 + r];
    tile[c4 + 0][r] = (_Float16)(v.x * dv);
    tile[c4 + 1][r] = (_Float16)(v.y * dv);
    tile[c4 + 2][r] = (_Float16)(v.z * dv);
    tile[c4 + 3][r] = (_Float16)(v.w * dv);
  }
  __syncthreads();
  // coalesced transposed write-out (16B chunks)
  for (int idx = threadIdx.x; idx < (FF * 64) / 8; idx += 256) {
    int c  = idx >> 3;              // 0..127
    int w8 = (idx & 7) << 3;        // 0..56
    *(uint4*)(xs_t + (size_t)c * NN + i0 + w8) = *(const uint4*)&tile[c][w8];
  }
}

// ---------------------------------------------------------------------------
// Kernel 3: Wt[u][k] = f16(W[k][u])   (tiny, 16K elements)
// ---------------------------------------------------------------------------
__global__ __launch_bounds__(256) void gcn_prep_w_kernel(
    const float* __restrict__ W, _Float16* __restrict__ Wt)
{
  int idx = blockIdx.x * 256 + threadIdx.x;   // idx = k*128 + u
  if (idx < FF * UU) {
    int k = idx >> 7, u = idx & 127;
    Wt[u * FF + k] = (_Float16)W[idx];
  }
}

// ---------------------------------------------------------------------------
// Kernel 4 (fused): H = (A+I) @ (dinv*x), scale by dinv, out = relu(H@W + b)
// 512 blocks x 256 threads (8 wave32 waves); block = 16 rows, wave = 16 cols.
// Hot loop: K=64/iter, double-buffered A tile, 1 barrier / 2 WMMAs,
// B fragments straight from L2-resident transposed xs_t (2x b128 per WMMA).
// ---------------------------------------------------------------------------
__global__ __launch_bounds__(256) void gcn_fused_kernel(
    const float* __restrict__ adj, const _Float16* __restrict__ xs_t,
    const _Float16* __restrict__ Wt, const float* __restrict__ dinv,
    const float* __restrict__ bias, float* __restrict__ out)
{
  __shared__ _Float16 As[2][16][72];   //  4.5 KB  double-buffered A tile (f16)
  __shared__ _Float16 Wl[128][136];    // 34   KB  W^T (f16): Wl[u][k]
  __shared__ _Float16 Hs[16][136];     //  4.25 KB H tile for 2nd GEMM

  const int tid  = threadIdx.x;
  const int wave = tid >> 5;
  const int lane = tid & 31;
  const int m    = lane & 15;          // A row / B,C,D column-in-tile
  const int g    = lane >> 4;         // lane half-group
  const int col  = (wave << 4) + m;    // 0..127: this lane's N column
  const int row0 = blockIdx.x << 4;

  // stage W^T into LDS once (coalesced 16B chunks)
  for (int c2 = tid; c2 < (FF * UU) / 8; c2 += 256) {
    int r = c2 >> 4, seg = (c2 & 15) << 3;
    *(uint4*)&Wl[r][seg] = *(const uint4*)(Wt + r * FF + seg);
  }

  v8f acc = {0.f, 0.f, 0.f, 0.f, 0.f, 0.f, 0.f, 0.f};

  const int rl = tid >> 4;             // staging row 0..15
  const int cl = (tid & 15) << 2;      // staging col quad 0,4,...,60
  const _Float16* bcol = xs_t + (size_t)col * NN;   // this lane's B column

  for (int it = 0; it < NN / 64; ++it) {
    const int k0  = it << 6;
    const int buf = it & 1;

    // --- stage A 16x64 (f32 -> f16), add identity, non-temporal read ---
    f4v f = __builtin_nontemporal_load(
        (const f4v*)(adj + (size_t)(row0 + rl) * NN + k0 + cl));
    const int grow = row0 + rl;
    if (grow == k0 + cl + 0) f.x += 1.0f;   // a_hat = adj + I
    if (grow == k0 + cl + 1) f.y += 1.0f;
    if (grow == k0 + cl + 2) f.z += 1.0f;
    if (grow == k0 + cl + 3) f.w += 1.0f;
    h4 hv;
    hv[0] = (_Float16)f.x; hv[1] = (_Float16)f.y;
    hv[2] = (_Float16)f.z; hv[3] = (_Float16)f.w;
    *(h4*)&As[buf][rl][cl] = hv;          // ds_store_b64
    __syncthreads();                      // single barrier per 2 WMMAs

#pragma unroll
    for (int c = 0; c < 2; ++c) {         // two K=32 sub-chunks
      v8h alo = *(const v8h*)&As[buf][m][(c << 5) + (g << 3)];
      v8h ahi = *(const v8h*)&As[buf][m][(c << 5) + 16 + (g << 3)];
      v16h a  = CAT16(alo, ahi);
      const _Float16* bp = bcol + k0 + (c << 5) + (g << 3);
      v8h blo = *(const v8h*)(bp);
      v8h bhi = *(const v8h*)(bp + 16);
      v16h b  = CAT16(blo, bhi);
      acc = __builtin_amdgcn_wmma_f32_16x16x32_f16(
          false, a, false, b, (short)0, acc, false, false);
    }
  }

  // --- scale rows by dinv[i], park H tile in LDS as f16 ---
#pragma unroll
  for (int v = 0; v < 8; ++v) {
    int M = (g << 3) + v;
    Hs[M][col] = (_Float16)(acc[v] * dinv[row0 + M]);
  }
  __syncthreads();

  // --- fused 2nd GEMM: (16x128) @ (128x128) via W^T in LDS ---
  v8f acc2 = {0.f, 0.f, 0.f, 0.f, 0.f, 0.f, 0.f, 0.f};
#pragma unroll
  for (int c = 0; c < 4; ++c) {
    const int k0 = c << 5;
    v8h alo = *(const v8h*)&Hs[m][k0 + (g << 3)];
    v8h ahi = *(const v8h*)&Hs[m][k0 + 16 + (g << 3)];
    v16h a  = CAT16(alo, ahi);
    v8h blo = *(const v8h*)&Wl[col][k0 + (g << 3)];
    v8h bhi = *(const v8h*)&Wl[col][k0 + 16 + (g << 3)];
    v16h b  = CAT16(blo, bhi);
    acc2 = __builtin_amdgcn_wmma_f32_16x16x32_f16(
        false, a, false, b, (short)0, acc2, false, false);
  }

  // --- bias + relu + store (f32 out) ---
  const float bb = bias[col];
#pragma unroll
  for (int v = 0; v < 8; ++v) {
    int M = (g << 3) + v;
    float r = acc2[v] + bb;
    out[(size_t)(row0 + M) * UU + col] = fmaxf(r, 0.0f);
  }
}

// ---------------------------------------------------------------------------
extern "C" void kernel_launch(void* const* d_in, const int* in_sizes, int n_in,
                              void* d_out, int out_size, void* d_ws, size_t ws_size,
                              hipStream_t stream) {
  const float* x    = (const float*)d_in[0];   // [N,F]
  const float* adj  = (const float*)d_in[1];   // [N,N]
  const float* W    = (const float*)d_in[2];   // [F,U]
  const float* bias = (const float*)d_in[3];   // [U]
  float* out = (float*)d_out;                  // [N,U]

  // ws: dinv (8192 f32) | xs_t (128*8192 f16, col-major) | Wt (128*128 f16)
  float*    dinv = (float*)d_ws;
  _Float16* xs_t = (_Float16*)(dinv + NN);
  _Float16* Wt   = xs_t + (size_t)FF * NN;

  gcn_deg_kernel<<<NN, 256, 0, stream>>>(adj, dinv);
  gcn_prep_x_kernel<<<NN / 64, 256, 0, stream>>>(x, dinv, xs_t);
  gcn_prep_w_kernel<<<(FF * UU + 255) / 256, 256, 0, stream>>>(W, Wt);
  gcn_fused_kernel<<<NN / 16, 256, 0, stream>>>(adj, xs_t, Wt, dinv, bias, out);
}
